// AttentionLayer_541165879329
// MI455X (gfx1250) — matrix operand order repaired and checked by reading, not compiled, not run on hardware
//
#include <hip/hip_runtime.h>
#include <hip/hip_bf16.h>

typedef __attribute__((ext_vector_type(16))) __bf16 v16bf;
typedef __attribute__((ext_vector_type(8)))  __bf16 v8bf;
typedef __attribute__((ext_vector_type(8)))  float  v8f;

#define DMODEL 1024
#define SEQ    2048
#define BATCH  4
#define HEADS  16
#define HDIM   64
#define MTOT   (BATCH*SEQ)   /* 8192 */

static __device__ __forceinline__ v8f bwmma(v16bf a, v16bf b, v8f c) {
  return __builtin_amdgcn_wmma_f32_16x16x32_bf16(false, a, false, b, (short)0, c, false, false);
}
static __device__ __forceinline__ v16bf ld16(const __bf16* p) {
  return *(const v16bf*)p;            // 32B vector load (32B-aligned by layout)
}
static __device__ __forceinline__ v16bf ld8x2(const __bf16* lo, const __bf16* hi) {
  v8bf a = *(const v8bf*)lo;          // two 16B chunks (A-fragment halves)
  v8bf b = *(const v8bf*)hi;
  v16bf r;
#pragma unroll
  for (int i = 0; i < 8; ++i) { r[i] = a[i]; r[i + 8] = b[i]; }
  return r;
}
// async copy of one 16B chunk from global to LDS (ASYNCcnt-tracked)
static __device__ __forceinline__ void async_b128(const __bf16* g, const __bf16* l) {
  unsigned      laddr = (unsigned)(uintptr_t)l;          // low 32b of generic = LDS offset
  unsigned long long gaddr = (unsigned long long)(uintptr_t)g;
  asm volatile("global_load_async_to_lds_b128 %0, %1, off"
               :: "v"(laddr), "v"(gaddr) : "memory");
}

// ---------------- elementwise f32 -> bf16 ----------------
__global__ void cvt_kernel(const float* __restrict__ in, __bf16* __restrict__ out, int n) {
  int i = blockIdx.x * blockDim.x + threadIdx.x;
  int stride = gridDim.x * blockDim.x;
  for (; i < n; i += stride) out[i] = (__bf16)in[i];
}

// ------------- W [k][n] f32 -> Wt [n][k] bf16 (LDS-tiled) -------------
__global__ void wtrans_kernel(const float* __restrict__ W, __bf16* __restrict__ Wt) {
  __shared__ float t[32][33];
  const int bx = blockIdx.x * 32, by = blockIdx.y * 32;
  const int x = threadIdx.x, y = threadIdx.y;      // blockDim = (32,8)
#pragma unroll
  for (int i = 0; i < 32; i += 8) t[y + i][x] = W[(size_t)(by + y + i) * DMODEL + bx + x];
  __syncthreads();
#pragma unroll
  for (int i = 0; i < 32; i += 8) Wt[(size_t)(bx + y + i) * DMODEL + by + x] = (__bf16)t[x][y + i];
}

// ------------- GEMM: out = A[8192][1024] @ W + bias -------------
// Wt is [n][k] bf16 (W transposed) so B-fragments are one contiguous 32B read.
// Each wave owns a 64x32 output tile: 8 accumulators; all 6 fragment loads of a
// K-step are issued ahead of the 8 WMMAs so the scheduler can clause them and
// retire WMMAs against partial s_wait_loadcnt values.
// mode 0: bf16 row-major, mode 1: bf16 transposed out[c*MTOT+m], mode 2: f32 row-major
__global__ void gemm_bf16_wmma(const __bf16* __restrict__ A,
                               const __bf16* __restrict__ Wt,
                               const float*  __restrict__ bias,
                               void* __restrict__ out, int mode) {
  const int lane   = threadIdx.x & 31;
  const int wave   = threadIdx.x >> 5;
  const int waveId = blockIdx.x * 8 + wave;        // 4096 waves
  const int mtile  = waveId >> 5;                  // 128 m-tiles of 64 rows
  const int ntile  = waveId & 31;                  // 32 n-tiles of 32 cols
  const int m0 = mtile * 64, n0 = ntile * 32;
  const int g = lane >> 4, ln = lane & 15;

  v8f acc[4][2] = {};
  const __bf16* brow0 = Wt + (size_t)(n0 + ln) * DMODEL + g * 16;
  const __bf16* brow1 = brow0 + (size_t)16 * DMODEL;
  const __bf16* arow[4];
#pragma unroll
  for (int j = 0; j < 4; ++j)
    arow[j] = A + (size_t)(m0 + j * 16 + ln) * DMODEL + g * 16;

  for (int k0 = 0; k0 < DMODEL; k0 += 32) {
    __builtin_prefetch((const void*)(brow0 + k0 + 128), 0, 3);
    __builtin_prefetch((const void*)(brow1 + k0 + 128), 0, 3);
    // issue ALL loads of this K-step first (12 x b128), then the 8 WMMAs
    v16bf b0 = ld16(brow0 + k0);
    v16bf b1 = ld16(brow1 + k0);
    v16bf a0 = ld16(arow[0] + k0);
    v16bf a1 = ld16(arow[1] + k0);
    v16bf a2 = ld16(arow[2] + k0);
    v16bf a3 = ld16(arow[3] + k0);
    acc[0][0] = bwmma(a0, b0, acc[0][0]);
    acc[0][1] = bwmma(a0, b1, acc[0][1]);
    acc[1][0] = bwmma(a1, b0, acc[1][0]);
    acc[1][1] = bwmma(a1, b1, acc[1][1]);
    acc[2][0] = bwmma(a2, b0, acc[2][0]);
    acc[2][1] = bwmma(a2, b1, acc[2][1]);
    acc[3][0] = bwmma(a3, b0, acc[3][0]);
    acc[3][1] = bwmma(a3, b1, acc[3][1]);
  }

  const float bv0 = bias[n0 + ln];
  const float bv1 = bias[n0 + 16 + ln];
#pragma unroll
  for (int j = 0; j < 4; ++j) {
#pragma unroll
    for (int c = 0; c < 2; ++c) {
      float bv = c ? bv1 : bv0;
#pragma unroll
      for (int r = 0; r < 8; ++r) {
        float v = acc[j][c][r] + bv;
        int row = m0 + j * 16 + r + g * 8;
        int col = n0 + c * 16 + ln;
        if (mode == 0)      ((__bf16*)out)[(size_t)row * DMODEL + col] = (__bf16)v;
        else if (mode == 1) ((__bf16*)out)[(size_t)col * MTOT + row]   = (__bf16)v;
        else                ((float*)out)[(size_t)row * DMODEL + col]  = v;
      }
    }
  }
}

// ------------- Flash attention -------------
// All 8 waves of a block share (batch nb, head h); K/V tiles for each 32-key
// block are staged once per block into LDS with async-to-LDS copies
// (double-buffered, ASYNCcnt), then consumed as WMMA B-fragments via ds reads.
// Qb,Kb: [8192][1024] bf16; Vt: [1024][8192] bf16 (dh-major); ctx: [8192][1024] bf16.
__global__ void attn_kernel(const __bf16* __restrict__ Qb,
                            const __bf16* __restrict__ Kb,
                            const __bf16* __restrict__ Vt,
                            __bf16* __restrict__ ctx) {
  __shared__ __align__(128) __bf16 ksh[2][32 * 64];   // [s][dh]  4KB per buffer
  __shared__ __align__(128) __bf16 vsh[2][64 * 32];   // [dh][s]  4KB per buffer
  __shared__ __align__(64)  __bf16 psh[8][16 * 32];   // per-wave P transpose tile

  const int tid    = threadIdx.x;
  const int lane   = tid & 31;
  const int wave   = tid >> 5;
  const int waveId = blockIdx.x * 8 + wave;          // 8192 = 4*16*128
  const int lt = waveId & 127;
  const int h  = (waveId >> 7) & 15;
  const int nb = waveId >> 11;
  const int l0 = lt * 16;
  const int g = lane >> 4, ln = lane & 15;
  const float scale = 0.125f;                        // 1/sqrt(64)
  __bf16* pw = &psh[wave][0];

  // per-thread 16B piece assignments for the staged tiles
  const int ks = tid >> 3, kc = tid & 7;             // K: 32 rows x 8 chunks
  const int vd = tid >> 2, vc = tid & 3;             // V: 64 rows x 4 chunks
  const __bf16* kgb = Kb + ((size_t)(nb * SEQ + ks)) * DMODEL + h * HDIM + kc * 8;
  const __bf16* vgb = Vt + ((size_t)(h * HDIM + vd)) * MTOT + nb * SEQ + vc * 8;

  auto stage = [&](int it) {
    const int buf = it & 1;
    const size_t s0 = (size_t)it * 32;
    async_b128(kgb + s0 * DMODEL, &ksh[buf][ks * 64 + kc * 8]);
    async_b128(vgb + s0,          &vsh[buf][vd * 32 + vc * 8]);
  };

  // Q A-fragments for Dh ranges [0,32) and [32,64) — loaded once, reused.
  const __bf16* qrow = Qb + ((size_t)(nb * SEQ + l0 + ln)) * DMODEL + h * HDIM;
  const v16bf qa0 = ld16(qrow + 0  + g * 16);
  const v16bf qa1 = ld16(qrow + 32 + g * 16);

  v8f o0 = {}, o1 = {}, o2 = {}, o3 = {};
  float mrow[8], lrow[8];
#pragma unroll
  for (int r = 0; r < 8; ++r) { mrow[r] = -3.0e38f; lrow[r] = 0.0f; }

  const int NIT = SEQ / 32;                          // 64
  stage(0);
  for (int it = 0; it < NIT; ++it) {
    if (it + 1 < NIT) {
      stage(it + 1);                                 // 2 new ops -> 4 outstanding
      asm volatile("s_wait_asynccnt 0x2" ::: "memory");  // current stage landed
    } else {
      asm volatile("s_wait_asynccnt 0x0" ::: "memory");
    }
    __syncthreads();                                 // tile visible block-wide

    const __bf16* kb = &ksh[it & 1][0];
    const __bf16* vb = &vsh[it & 1][0];

    // S = Q @ K^T : B-frag lane = key (LDS rows contiguous in dh)
    v8f sA = {}, sB = {};
    sA = bwmma(qa0, ld16(kb + (0  + ln) * 64 + 0  + g * 16), sA);
    sA = bwmma(qa1, ld16(kb + (0  + ln) * 64 + 32 + g * 16), sA);
    sB = bwmma(qa0, ld16(kb + (16 + ln) * 64 + 0  + g * 16), sB);
    sB = bwmma(qa1, ld16(kb + (16 + ln) * 64 + 32 + g * 16), sB);

    // online softmax: rows live in VGPR index r across a 16-lane group
#pragma unroll
    for (int r = 0; r < 8; ++r) {
      float x0 = sA[r] * scale, x1 = sB[r] * scale;
      float tm = fmaxf(x0, x1);
#pragma unroll
      for (int m = 1; m < 16; m <<= 1) tm = fmaxf(tm, __shfl_xor(tm, m, 32));
      float mn = fmaxf(mrow[r], tm);
      float p0 = __expf(x0 - mn), p1 = __expf(x1 - mn);
      float rs = p0 + p1;
#pragma unroll
      for (int m = 1; m < 16; m <<= 1) rs += __shfl_xor(rs, m, 32);
      float alpha = __expf(mrow[r] - mn);
      lrow[r] = lrow[r] * alpha + rs;
      mrow[r] = mn;
      o0[r] *= alpha; o1[r] *= alpha; o2[r] *= alpha; o3[r] *= alpha;
      const int prow = r + g * 8;                    // stage P (C-layout) as bf16
      pw[prow * 32 + ln]      = (__bf16)p0;
      pw[prow * 32 + 16 + ln] = (__bf16)p1;
    }
    // re-read P in A-fragment layout (per-wave LDS ops in-order: RAW safe)
    const __bf16* pr = pw + ln * 32 + g * 16;
    v16bf pa = ld16(pr);

    // O += P @ V : V tile rows are dh, contiguous in key position
    o0 = bwmma(pa, ld16(vb + ( 0 + ln) * 32 + g * 16), o0);
    o1 = bwmma(pa, ld16(vb + (16 + ln) * 32 + g * 16), o1);
    o2 = bwmma(pa, ld16(vb + (32 + ln) * 32 + g * 16), o2);
    o3 = bwmma(pa, ld16(vb + (48 + ln) * 32 + g * 16), o3);

    __syncthreads();                                 // all reads done before buffer reuse
  }

#pragma unroll
  for (int r = 0; r < 8; ++r) {
    float inv = 1.0f / lrow[r];
    int row = l0 + r + g * 8;
    size_t base = ((size_t)(nb * SEQ + row)) * DMODEL + h * HDIM + ln;
    ctx[base + 0]  = (__bf16)(o0[r] * inv);
    ctx[base + 16] = (__bf16)(o1[r] * inv);
    ctx[base + 32] = (__bf16)(o2[r] * inv);
    ctx[base + 48] = (__bf16)(o3[r] * inv);
  }
}

extern "C" void kernel_launch(void* const* d_in, const int* in_sizes, int n_in,
                              void* d_out, int out_size, void* d_ws, size_t ws_size,
                              hipStream_t stream) {
  (void)in_sizes; (void)n_in; (void)out_size; (void)ws_size;
  const float* x  = (const float*)d_in[0];
  const float* Wq = (const float*)d_in[1];
  const float* bq = (const float*)d_in[2];
  const float* Wk = (const float*)d_in[3];
  const float* bk = (const float*)d_in[4];
  const float* Wv = (const float*)d_in[5];
  const float* bv = (const float*)d_in[6];
  const float* Wo = (const float*)d_in[7];
  const float* bo = (const float*)d_in[8];

  char* ws = (char*)d_ws;
  size_t off = 0;
  __bf16* xb   = (__bf16*)(ws + off); off += (size_t)MTOT * DMODEL * 2;   // 16 MB
  __bf16* wtq  = (__bf16*)(ws + off); off += (size_t)DMODEL * DMODEL * 2; //  2 MB
  __bf16* wtk  = (__bf16*)(ws + off); off += (size_t)DMODEL * DMODEL * 2;
  __bf16* wtv  = (__bf16*)(ws + off); off += (size_t)DMODEL * DMODEL * 2;
  __bf16* wto  = (__bf16*)(ws + off); off += (size_t)DMODEL * DMODEL * 2;
  __bf16* qb   = (__bf16*)(ws + off); off += (size_t)MTOT * DMODEL * 2;
  __bf16* kbuf = (__bf16*)(ws + off); off += (size_t)MTOT * DMODEL * 2;
  __bf16* vt   = (__bf16*)(ws + off); off += (size_t)MTOT * DMODEL * 2;   // [1024][8192]
  __bf16* ctx  = (__bf16*)(ws + off); off += (size_t)MTOT * DMODEL * 2;

  // 1) precision conversion
  cvt_kernel<<<4096, 256, 0, stream>>>(x, xb, MTOT * DMODEL);
  dim3 tb(32, 8), tg(32, 32);
  wtrans_kernel<<<tg, tb, 0, stream>>>(Wq, wtq);
  wtrans_kernel<<<tg, tb, 0, stream>>>(Wk, wtk);
  wtrans_kernel<<<tg, tb, 0, stream>>>(Wv, wtv);
  wtrans_kernel<<<tg, tb, 0, stream>>>(Wo, wto);

  // 2) projections (V stored transposed for PV B-fragments)
  gemm_bf16_wmma<<<512, 256, 0, stream>>>(xb, wtq, bq, (void*)qb,   0);
  gemm_bf16_wmma<<<512, 256, 0, stream>>>(xb, wtk, bk, (void*)kbuf, 0);
  gemm_bf16_wmma<<<512, 256, 0, stream>>>(xb, wtv, bv, (void*)vt,   1);

  // 3) attention (async-staged K/V tiles in LDS)
  attn_kernel<<<1024, 256, 0, stream>>>(qb, kbuf, vt, ctx);

  // 4) output projection (f32 out + bias)
  gemm_bf16_wmma<<<512, 256, 0, stream>>>(ctx, wto, bo, d_out, 2);
}
